// MyRNN_skip_84490596647084
// MI455X (gfx1250) — matrix operand order repaired and checked by reading, not compiled, not run on hardware
//
#include <hip/hip_runtime.h>

#define B_   64
#define T_   512
#define FIN_ 256
#define H_   1024
#define OUT_ 64
#define DIN_ 384   // FIN_ + 2*OUT_

typedef __attribute__((ext_vector_type(16))) __bf16 bf16x16;
typedef __attribute__((ext_vector_type(8)))  float  f32x8;

#define WMMA_BF16(a, b, c) \
  __builtin_amdgcn_wmma_f32_16x16x32_bf16(false, (a), false, (b), (short)0, (c), false, false)

__device__ __forceinline__ float sigmoidf_(float x) { return 1.0f / (1.0f + __expf(-x)); }

// A fragment: 16x32 bf16 tile from row-major LDS, leading dim ld (elements).
// ISA layout: lanes 0-15 -> M=lane, K in {0..7,16..23}; lanes 16-31 -> M=lane-16, K in {8..15,24..31}.
// Lowers to two ds_load_b128.
__device__ __forceinline__ bf16x16 load_A(const __bf16* base, int ld) {
  const int lane = threadIdx.x & 31;
  const int row  = lane & 15;
  const int kb   = (lane >> 4) * 8;
  const __bf16* p = base + (size_t)row * ld + kb;
  bf16x16 a;
#pragma unroll
  for (int v = 0; v < 4; ++v) { a[2*v] = p[2*v]; a[2*v+1] = p[2*v+1]; }
#pragma unroll
  for (int v = 0; v < 4; ++v) { a[8+2*v] = p[16+2*v]; a[8+2*v+1] = p[16+2*v+1]; }
  return a;
}

// Packed-weight B fragment: each lane's 16 bf16 fragment elements are
// contiguous (32B @ lane*32 within a 1KB tile); tiles ordered [n_tile][k_step].
// Lowers to two global_load_b128, fully coalesced (wave reads 1KB sequential).
__device__ __forceinline__ bf16x16 load_B_packed(const __bf16* __restrict__ Wp,
                                                 int ksteps, int ks, int ntile) {
  const int lane = threadIdx.x & 31;
  const __bf16* p = Wp + (((size_t)ntile * ksteps + ks) << 9) + (lane << 4);
  return *(const bf16x16*)p;
}

// One 16x16 output tile over K = KSTEPS*32.
template <int KSTEPS>
__device__ __forceinline__ f32x8 gemm_row(const __bf16* A, int lda,
                                          const __bf16* __restrict__ Wp, int ntile) {
  f32x8 acc = {0.f, 0.f, 0.f, 0.f, 0.f, 0.f, 0.f, 0.f};
  const __bf16* tp = Wp + ((size_t)ntile * KSTEPS << 9) + ((threadIdx.x & 31) << 4);
#pragma unroll 2
  for (int ks = 0; ks < KSTEPS; ++ks) {
    bf16x16 a = load_A(A + ks * 32, lda);
    bf16x16 b = *(const bf16x16*)(tp + ((size_t)ks << 9));
    acc = WMMA_BF16(a, b, acc);
  }
  return acc;
}

// Four 16x16 output tiles (ntile = jt_base + u*jt_step) sharing one A-fragment
// load per K-step. All four B fragments are loaded into distinct registers
// BEFORE the four WMMAs so the loads clause together and are consumed with
// partial loadcnt waits, overlapping VMEM with the matrix pipe.
template <int KSTEPS>
__device__ __forceinline__ void gemm_row4(const __bf16* A, int lda,
                                          const __bf16* __restrict__ Wp,
                                          int jt_base, int jt_step, f32x8 acc[4]) {
  const int lane16 = (threadIdx.x & 31) << 4;
  const __bf16* tp[4];
#pragma unroll
  for (int u = 0; u < 4; ++u) {
    tp[u] = Wp + ((size_t)(jt_base + u * jt_step) * KSTEPS << 9) + lane16;
#pragma unroll
    for (int v = 0; v < 8; ++v) acc[u][v] = 0.f;
  }
#pragma unroll 2
  for (int ks = 0; ks < KSTEPS; ++ks) {
    const size_t o = (size_t)ks << 9;
    bf16x16 a  = load_A(A + ks * 32, lda);
    bf16x16 b0 = *(const bf16x16*)(tp[0] + o);
    bf16x16 b1 = *(const bf16x16*)(tp[1] + o);
    bf16x16 b2 = *(const bf16x16*)(tp[2] + o);
    bf16x16 b3 = *(const bf16x16*)(tp[3] + o);
    acc[0] = WMMA_BF16(a, b0, acc[0]);
    acc[1] = WMMA_BF16(a, b1, acc[1]);
    acc[2] = WMMA_BF16(a, b2, acc[2]);
    acc[3] = WMMA_BF16(a, b3, acc[3]);
  }
}

// C/D layout: VGPR v, lanes 0-15: (M=v, N=lane); lanes 16-31: (M=8+v, N=lane-16).
__device__ __forceinline__ void store_relu(__bf16* base, int ld, f32x8 acc,
                                           const float* bias, int n0) {
  const int lane  = threadIdx.x & 31;
  const int n     = n0 + (lane & 15);
  const float bb  = bias[n];
  const int mbase = (lane >> 4) * 8;
#pragma unroll
  for (int v = 0; v < 8; ++v) {
    float xv = acc[v] + bb;
    base[(size_t)(mbase + v) * ld + n] = (__bf16)fmaxf(xv, 0.f);
  }
}

__global__ __launch_bounds__(512, 1)
void rnn_persist(const float* __restrict__ x, const int* __restrict__ x_len,
                 const float* __restrict__ y0v, const float* __restrict__ y1v,
                 const float* __restrict__ b_e_in, const float* __restrict__ b_e_h,
                 const float* __restrict__ b_e_out,
                 const float* __restrict__ b_ih, const float* __restrict__ b_hh,
                 const float* __restrict__ b_o_in, const float* __restrict__ b_o_h,
                 const float* __restrict__ b_o_out,
                 const __bf16* __restrict__ w_e_in, const __bf16* __restrict__ w_e_h,
                 const __bf16* __restrict__ w_e_out,
                 const __bf16* __restrict__ w_ih, const __bf16* __restrict__ w_hh,
                 const __bf16* __restrict__ w_o_in, const __bf16* __restrict__ w_o_h,
                 const __bf16* __restrict__ w_o_out,
                 float* __restrict__ out) {
  // ~184 KB static LDS (gfx1250 WGP has 320 KB)
  __shared__ __bf16 s_z [16 * DIN_];
  __shared__ __bf16 s_b0[16 * H_];
  __shared__ __bf16 s_b1[16 * H_];
  __shared__ __bf16 s_r [16 * H_];
  __shared__ __bf16 s_hA[16 * H_];
  __shared__ __bf16 s_hB[16 * H_];
  __shared__ float  s_y [16 * OUT_];
  __shared__ float  s_yp[16 * OUT_];
  __shared__ int    s_xlen[16];

  const int tid   = threadIdx.x;
  const int w     = tid >> 5;
  const int lane  = tid & 31;
  const int bbase = blockIdx.x * 16;

  for (int i = tid; i < 16 * H_; i += 512) s_hA[i] = (__bf16)0.f;
  for (int i = tid; i < 16 * OUT_; i += 512) {
    int m = i >> 6, n = i & 63;
    s_y[i]  = y1v[(size_t)(bbase + m) * OUT_ + n];
    s_yp[i] = y0v[(size_t)(bbase + m) * OUT_ + n];
  }
  if (tid < 16) s_xlen[tid] = x_len[bbase + tid];

  f32x8 c_reg[4];
#pragma unroll
  for (int q = 0; q < 4; ++q)
#pragma unroll
    for (int v = 0; v < 8; ++v) c_reg[q][v] = 0.f;
  __syncthreads();

  float* ys_out = out;                              // (B,T,OUT)
  float* sk_out = out + (size_t)B_ * T_ * OUT_;     // (B,T,H)

#pragma unroll 1
  for (int t = 0; t < T_; ++t) {
    const __bf16* h_cur = (t & 1) ? s_hB : s_hA;
    __bf16*       h_nxt = (t & 1) ? s_hA : s_hB;

    // z = [x_t | y | y_prev]  (bf16, 16 x 384)
    for (int i = tid; i < 16 * FIN_; i += 512) {
      int m = i >> 8, n = i & 255;
      s_z[m * DIN_ + n] = (__bf16)x[((size_t)(bbase + m) * T_ + t) * FIN_ + n];
    }
    for (int i = tid; i < 16 * OUT_; i += 512) {
      int m = i >> 6, n = i & 63;
      s_z[m * DIN_ + FIN_ + n]        = (__bf16)s_y[i];
      s_z[m * DIN_ + FIN_ + OUT_ + n] = (__bf16)s_yp[i];
    }
    __syncthreads();

    // y_prev <- y (z already captured both)
    for (int i = tid; i < 16 * OUT_; i += 512) s_yp[i] = s_y[i];

    // Encoder L1: a1 = relu(z @ W_e_in + b)
    {
      f32x8 acc[4];
      gemm_row4<12>(s_z, DIN_, w_e_in, w, 16, acc);
#pragma unroll
      for (int u = 0; u < 4; ++u)
        store_relu(s_b0, H_, acc[u], b_e_in, (w + 16 * u) * 16);
    }
    __syncthreads();

    // Encoder L2: a2 = relu(a1 @ W_e_h + b)
    {
      f32x8 acc[4];
      gemm_row4<32>(s_b0, H_, w_e_h, w, 16, acc);
#pragma unroll
      for (int u = 0; u < 4; ++u)
        store_relu(s_b1, H_, acc[u], b_e_h, (w + 16 * u) * 16);
    }
    __syncthreads();

    // Encoder L3: e = a2 @ W_e_out + b (skip output, pre-relu) ; r = relu(e)
    {
      f32x8 acc[4];
      gemm_row4<32>(s_b1, H_, w_e_out, w, 16, acc);
      const int n     = lane & 15;
      const int mbase = (lane >> 4) * 8;
#pragma unroll
      for (int u = 0; u < 4; ++u) {
        const int jt   = w + 16 * u;
        const float bb = b_e_out[jt * 16 + n];
#pragma unroll
        for (int v = 0; v < 8; ++v) {
          int m   = mbase + v;
          float e = acc[u][v] + bb;
          float mask = (t < s_xlen[m]) ? 1.f : 0.f;
          sk_out[((size_t)(bbase + m) * T_ + t) * H_ + jt * 16 + n] = e * mask;
          s_r[(size_t)m * H_ + jt * 16 + n] = (__bf16)fmaxf(e, 0.f);
        }
      }
    }
    __syncthreads();

    // LSTM: gates = r@W_ih + h@W_hh + b ; fused elementwise cell update.
    // Wave w permanently owns H-tiles w*4..w*4+3 -> cell state c lives in VGPRs.
    // Per K-step: 2 A fragments + 8 B fragments loaded up-front (distinct regs,
    // clause + partial waits), then 8 WMMAs; the i-gate RAW pair is separated
    // by 3 intervening WMMAs (hazard distance covered).
#pragma unroll 1
    for (int q = 0; q < 4; ++q) {
      const int jt = w * 4 + q;
      const int n0 = jt * 16;
      f32x8 ai = {0.f,0.f,0.f,0.f,0.f,0.f,0.f,0.f};
      f32x8 af = ai, ag = ai, ao = ai;
#pragma unroll 1
      for (int ks = 0; ks < 32; ++ks) {
        bf16x16 aR  = load_A(s_r + ks * 32, H_);
        bf16x16 aH  = load_A(h_cur + ks * 32, H_);
        bf16x16 bRi = load_B_packed(w_ih, 32, ks, 0 * 64 + jt);
        bf16x16 bRf = load_B_packed(w_ih, 32, ks, 1 * 64 + jt);
        bf16x16 bRg = load_B_packed(w_ih, 32, ks, 2 * 64 + jt);
        bf16x16 bRo = load_B_packed(w_ih, 32, ks, 3 * 64 + jt);
        bf16x16 bHi = load_B_packed(w_hh, 32, ks, 0 * 64 + jt);
        bf16x16 bHf = load_B_packed(w_hh, 32, ks, 1 * 64 + jt);
        bf16x16 bHg = load_B_packed(w_hh, 32, ks, 2 * 64 + jt);
        bf16x16 bHo = load_B_packed(w_hh, 32, ks, 3 * 64 + jt);
        ai = WMMA_BF16(aR, bRi, ai);
        af = WMMA_BF16(aR, bRf, af);
        ag = WMMA_BF16(aR, bRg, ag);
        ao = WMMA_BF16(aR, bRo, ao);
        ai = WMMA_BF16(aH, bHi, ai);
        af = WMMA_BF16(aH, bHf, af);
        ag = WMMA_BF16(aH, bHg, ag);
        ao = WMMA_BF16(aH, bHo, ao);
      }
      const int n     = lane & 15;
      const int mbase = (lane >> 4) * 8;
      const float bi = b_ih[0 * H_ + n0 + n] + b_hh[0 * H_ + n0 + n];
      const float bf = b_ih[1 * H_ + n0 + n] + b_hh[1 * H_ + n0 + n];
      const float bg = b_ih[2 * H_ + n0 + n] + b_hh[2 * H_ + n0 + n];
      const float bo = b_ih[3 * H_ + n0 + n] + b_hh[3 * H_ + n0 + n];
#pragma unroll
      for (int v = 0; v < 8; ++v) {
        int m = mbase + v;
        float I = sigmoidf_(ai[v] + bi);
        float F = sigmoidf_(af[v] + bf);
        float G = tanhf(ag[v] + bg);
        float O = sigmoidf_(ao[v] + bo);
        float cn = F * c_reg[q][v] + I * G;
        float hn = O * tanhf(cn);
        c_reg[q][v] = cn;
        h_nxt[(size_t)m * H_ + n0 + n] = (__bf16)hn;
        s_b0[(size_t)m * H_ + n0 + n]  = (__bf16)fmaxf(hn, 0.f);  // q = relu(h_new)
      }
    }
    __syncthreads();

    // Decoder L1: o1 = relu(q @ W_o_in + b)
    {
      f32x8 acc[4];
      gemm_row4<32>(s_b0, H_, w_o_in, w, 16, acc);
#pragma unroll
      for (int u = 0; u < 4; ++u)
        store_relu(s_r, H_, acc[u], b_o_in, (w + 16 * u) * 16);
    }
    __syncthreads();

    // Decoder L2: o2 = relu(o1 @ W_o_h + b)
    {
      f32x8 acc[4];
      gemm_row4<32>(s_r, H_, w_o_h, w, 16, acc);
#pragma unroll
      for (int u = 0; u < 4; ++u)
        store_relu(s_b1, H_, acc[u], b_o_h, (w + 16 * u) * 16);
    }
    __syncthreads();

    // Decoder L3: y = o2 @ W_o_out + b  (OUT=64 -> 4 tiles on waves 0..3)
    if (w < 4) {
      const int jt = w;
      f32x8 acc = gemm_row<32>(s_b1, H_, w_o_out, jt);
      const int n     = lane & 15;
      const float bb  = b_o_out[jt * 16 + n];
      const int mbase = (lane >> 4) * 8;
#pragma unroll
      for (int v = 0; v < 8; ++v) {
        int m    = mbase + v;
        float yv = acc[v] + bb;
        s_y[m * OUT_ + jt * 16 + n] = yv;
        float mask = (t < s_xlen[m]) ? 1.f : 0.f;
        ys_out[((size_t)(bbase + m) * T_ + t) * OUT_ + jt * 16 + n] = yv * mask;
      }
    }
    __syncthreads();
  }
}

// Repack a row-major KxN f32 weight matrix into bf16 WMMA-fragment tile order:
// tile index = n_tile*(K/32) + k_step; within a 1KB tile, lane L's 16 elements
// are W[k_step*32 + (L>>4)*8 + (e&7) + (e>>3)*16, n_tile*16 + (L&15)], e = 0..15, contiguous.
__global__ void pack_w_bf16(const float* __restrict__ src, __bf16* __restrict__ dst,
                            int K, int N) {
  int p = blockIdx.x * blockDim.x + threadIdx.x;
  int total = K * N;
  if (p >= total) return;
  int KS   = K >> 5;
  int tile = p >> 9;
  int q    = p & 511;
  int lane = q >> 4;
  int e    = q & 15;
  int nt   = tile / KS;
  int ks   = tile - nt * KS;
  int col  = nt * 16 + (lane & 15);
  int k    = ks * 32 + ((lane >> 4) << 3) + (e & 7) + ((e >> 3) << 4);
  dst[p] = (__bf16)src[(size_t)k * N + col];
}

extern "C" void kernel_launch(void* const* d_in, const int* in_sizes, int n_in,
                              void* d_out, int out_size, void* d_ws, size_t ws_size,
                              hipStream_t stream) {
  const float* x       = (const float*)d_in[0];
  const int*   xlen    = (const int*)d_in[1];
  const float* y0v     = (const float*)d_in[2];
  const float* y1v     = (const float*)d_in[3];
  const float* W_e_in  = (const float*)d_in[4];
  const float* b_e_in  = (const float*)d_in[5];
  const float* W_e_h   = (const float*)d_in[6];
  const float* b_e_h   = (const float*)d_in[7];
  const float* W_e_out = (const float*)d_in[8];
  const float* b_e_out = (const float*)d_in[9];
  const float* W_ih    = (const float*)d_in[10];
  const float* W_hh    = (const float*)d_in[11];
  const float* b_ih    = (const float*)d_in[12];
  const float* b_hh    = (const float*)d_in[13];
  const float* W_o_in  = (const float*)d_in[14];
  const float* b_o_in  = (const float*)d_in[15];
  const float* W_o_h   = (const float*)d_in[16];
  const float* b_o_h   = (const float*)d_in[17];
  const float* W_o_out = (const float*)d_in[18];
  const float* b_o_out = (const float*)d_in[19];
  (void)in_sizes; (void)n_in; (void)out_size; (void)ws_size;

  __bf16* ws  = (__bf16*)d_ws;
  size_t  off = 0;
  auto pack = [&](const float* src, int K, int N) -> __bf16* {
    __bf16* dst = ws + off;
    size_t n = (size_t)K * N;
    off += n;
    int blocks = (int)((n + 255) / 256);
    pack_w_bf16<<<blocks, 256, 0, stream>>>(src, dst, K, N);
    return dst;
  };
  __bf16* we_in  = pack(W_e_in,  DIN_, H_);
  __bf16* we_h   = pack(W_e_h,   H_, H_);
  __bf16* we_out = pack(W_e_out, H_, H_);
  __bf16* wih    = pack(W_ih,    H_, 4 * H_);
  __bf16* whh    = pack(W_hh,    H_, 4 * H_);
  __bf16* wo_in  = pack(W_o_in,  H_, H_);
  __bf16* wo_h   = pack(W_o_h,   H_, H_);
  __bf16* wo_out = pack(W_o_out, H_, OUT_);

  rnn_persist<<<B_ / 16, 512, 0, stream>>>(
      x, xlen, y0v, y1v,
      b_e_in, b_e_h, b_e_out, b_ih, b_hh, b_o_in, b_o_h, b_o_out,
      we_in, we_h, we_out, wih, whh, wo_in, wo_h, wo_out,
      (float*)d_out);
}
